// PhysNetCore_42880953484208
// MI455X (gfx1250) — matrix operand order repaired and checked by reading, not compiled, not run on hardware
//
#include <hip/hip_runtime.h>
#include <hip/hip_bf16.h>

// ---------------------------------------------------------------------------
// PhysNet core on MI455X (gfx1250, wave32).
// - All K=128 GEMMs: V_WMMA_F32_16X16X32_F16 (f16 operands, f32 accum).
// - Edge RBF gate (f_ij @ Wg, K=16): V_WMMA_F32_16X16X4_F32 (native f32 WMMA).
// - Weights pre-transposed to f16 W^T[n][k]; staged 32KB/WG in LDS so each
//   lane's B-fragment is one contiguous 32B run.
// - softplus(emb) stored once as f16 (halves the 600k-edge gather traffic);
//   residual mid-activations stored f16 (only ever consumed as GEMM A-input).
// - softplus uses hw transcendentals (v_exp_f32/v_log_f32), not libm log1pf.
// ---------------------------------------------------------------------------

typedef __attribute__((ext_vector_type(16))) _Float16 v16h;
typedef __attribute__((ext_vector_type(8)))  _Float16 v8h;
typedef __attribute__((ext_vector_type(8)))  float    v8f;
typedef __attribute__((ext_vector_type(2)))  float    v2f;

#define F_DIM 128

__device__ __forceinline__ float softplus_f(float x) {
    // stable softplus on hw transcendentals: max(x,0) + log(1+exp(-|x|))
    return fmaxf(x, 0.0f) + __logf(1.0f + __expf(-fabsf(x)));
}

// ---------------- prep: sp = softplus(emb) in f32 and f16 -------------------
__global__ __launch_bounds__(256)
void softplus_dual_kernel(const float* __restrict__ in, float* __restrict__ out32,
                          _Float16* __restrict__ out16, size_t n) {
    size_t i = (size_t)blockIdx.x * blockDim.x + threadIdx.x;
    size_t stride = (size_t)gridDim.x * blockDim.x;
    for (; i < n; i += stride) {
        float s = softplus_f(in[i]);
        out32[i] = s;
        out16[i] = (_Float16)s;
    }
}

// ---------------- weight convert+transpose: W[k][n] -> f16 WT[n][k] ---------
__global__ __launch_bounds__(256)
void wt_transpose_kernel(const float* __restrict__ src, _Float16* __restrict__ dst) {
    int mat = blockIdx.x;
    const float* s = src + (size_t)mat * F_DIM * F_DIM;
    _Float16*    d = dst + (size_t)mat * F_DIM * F_DIM;
    for (int i = threadIdx.x; i < F_DIM * F_DIM; i += blockDim.x) {
        int k = i >> 7, n = i & 127;
        d[n * F_DIM + k] = (_Float16)s[i];
    }
}

// ---------------------------------------------------------------------------
// Generic node GEMM: Y = post( pre(X) @ W + b [, addsrc, gate] )
// PRE : 0 = identity, 1 = softplus applied to X on load (f32 path only)
// AH  : 0 = X is f32, 1 = X is f16 (already activated)
// POST: 1 = softplus(acc+b)            -> f32 Y
//       2 = addsrc[row] + acc + b      -> f32 Y (residual tail)
//       3 = gate[n]*addsrc[row]+acc+b  -> f32 Y (v = gate*e + u@Wv+bv)
//       4 = softplus(acc+b)            -> f16 Y (GEMM-input-only temp)
// Workgroup: 256 threads = 8 waves; each wave owns 16 rows x 128 cols.
// ---------------------------------------------------------------------------
template <int PRE, int POST, int AH>
__global__ __launch_bounds__(256)
void node_gemm_kernel(const void* __restrict__ Xv,
                      const _Float16* __restrict__ WT,   // [128 n][128 k] f16
                      const float* __restrict__ bias,    // [128]
                      const float* __restrict__ addsrc,  // [N,128] or null
                      const float* __restrict__ gate,    // [128] or null
                      void* __restrict__ Yv,
                      int nRows) {
    __shared__ _Float16 sW[F_DIM * F_DIM] __attribute__((aligned(32)));

    // Cooperative 32KB weight stage into LDS (uint4 granularity).
    {
        const uint4* s = (const uint4*)WT;
        uint4*       d = (uint4*)sW;
        for (int i = threadIdx.x; i < (F_DIM * F_DIM) / 8; i += 256) d[i] = s[i];
    }

    const int lane = threadIdx.x & 31;
    const int wave = threadIdx.x >> 5;
    const int mrow = lane & 15;        // A row / D col
    const int hi   = lane >> 4;        // half-wave selector

    const int rowBase = blockIdx.x * 128 + wave * 16;
    int rowA = rowBase + mrow;
    int rowAc = (rowA < nRows) ? rowA : (nRows - 1);

    // Build all four A fragments (K chunks of 32) once; reuse over 8 N tiles.
    v16h afrag[4];
    if (AH) {
        const _Float16* xr = (const _Float16*)Xv + (size_t)rowAc * F_DIM;
        __builtin_prefetch(xr, 0, 3);
        __syncthreads();
#pragma unroll
        for (int kt = 0; kt < 4; ++kt) {
            const int koff = kt * 32 + hi * 16;
            const v8h lo = *(const v8h*)&xr[koff];
            const v8h hh = *(const v8h*)&xr[koff + 8];
#pragma unroll
            for (int i = 0; i < 8; ++i) {
                afrag[kt][i]     = lo[i];
                afrag[kt][8 + i] = hh[i];
            }
        }
    } else {
        const float* xr = (const float*)Xv + (size_t)rowAc * F_DIM;
        __builtin_prefetch(xr, 0, 3);
        __syncthreads();
#pragma unroll
        for (int kt = 0; kt < 4; ++kt) {
            const int koff = kt * 32 + hi * 16;
#pragma unroll
            for (int i = 0; i < 8; ++i) {
                float v0 = xr[koff + i];
                float v1 = xr[koff + 8 + i];
                if (PRE == 1) { v0 = softplus_f(v0); v1 = softplus_f(v1); }
                afrag[kt][i]     = (_Float16)v0;
                afrag[kt][8 + i] = (_Float16)v1;
            }
        }
    }

#pragma unroll
    for (int nt = 0; nt < 8; ++nt) {
        const int n = nt * 16 + mrow;
        v8f c = {};
#pragma unroll
        for (int kt = 0; kt < 4; ++kt) {
            const v16h b = *(const v16h*)&sW[n * F_DIM + kt * 32 + hi * 16];
            c = __builtin_amdgcn_wmma_f32_16x16x32_f16(
                    false, afrag[kt], false, b, (short)0, c, false, false);
        }
        const float bn = bias[n];
#pragma unroll
        for (int r = 0; r < 8; ++r) {
            const int row = rowBase + r + hi * 8;
            if (row < nRows) {
                float acc = c[r] + bn;
                if (POST == 1) {
                    ((float*)Yv)[(size_t)row * F_DIM + n] = softplus_f(acc);
                } else if (POST == 2) {
                    acc += addsrc[(size_t)row * F_DIM + n];
                    ((float*)Yv)[(size_t)row * F_DIM + n] = acc;
                } else if (POST == 3) {
                    acc += gate[n] * addsrc[(size_t)row * F_DIM + n];
                    ((float*)Yv)[(size_t)row * F_DIM + n] = acc;
                } else { // POST == 4
                    ((_Float16*)Yv)[(size_t)row * F_DIM + n] = (_Float16)softplus_f(acc);
                }
            }
        }
    }
}

// ---------------------------------------------------------------------------
// Edge kernel: m_j = softplus(sph[idx_j] @ Wj + bj) * (f_ij @ Wg)
//              u[idx_i] += m_j   (f32 global atomics)
// 128 edges per workgroup (8 waves x 16 edges).
// Main GEMM on f16 WMMA; RBF gate on f32 WMMA (16x16x4, K=16 -> 4 ops).
// ---------------------------------------------------------------------------
__global__ __launch_bounds__(256)
void edge_kernel(const _Float16* __restrict__ sph,  // [N,128] f16 softplus(emb)
                 const float* __restrict__ f_ij,    // [E,16]
                 const int*   __restrict__ idx_i,
                 const int*   __restrict__ idx_j,
                 const _Float16* __restrict__ WjT,
                 const float* __restrict__ bj,
                 const float* __restrict__ Wg,      // [16,128]
                 float* __restrict__ u,
                 int nEdges) {
    __shared__ _Float16 sW[F_DIM * F_DIM] __attribute__((aligned(32)));
    __shared__ float sWg[16 * F_DIM] __attribute__((aligned(16)));
    __shared__ float sF[128 * 16] __attribute__((aligned(16)));
    __shared__ int   sIi[128];

    {
        const uint4* s = (const uint4*)WjT;
        uint4*       d = (uint4*)sW;
        for (int i = threadIdx.x; i < (F_DIM * F_DIM) / 8; i += 256) d[i] = s[i];
    }
    for (int i = threadIdx.x; i < 16 * F_DIM; i += 256) sWg[i] = Wg[i];

    const int e0 = blockIdx.x * 128;
    for (int i = threadIdx.x; i < 128 * 16; i += 256) {
        int e = e0 + (i >> 4);
        sF[i] = (e < nEdges) ? f_ij[(size_t)e * 16 + (i & 15)] : 0.0f;
    }
    for (int i = threadIdx.x; i < 128; i += 256) {
        int e = e0 + i;
        sIi[i] = (e < nEdges) ? idx_i[e] : -1;
    }

    const int lane = threadIdx.x & 31;
    const int wave = threadIdx.x >> 5;
    const int mrow = lane & 15;
    const int hi   = lane >> 4;

    const int edgeA  = e0 + wave * 16 + mrow;
    const int edgeAc = (edgeA < nEdges) ? edgeA : (nEdges - 1);
    const int jrow   = idx_j[edgeAc];
    const _Float16* xr = sph + (size_t)jrow * F_DIM;
    __builtin_prefetch(xr, 0, 3);   // global_prefetch_b8

    __syncthreads();

    // f16 A fragments for the message GEMM: two contiguous 16B runs per chunk.
    v16h afrag[4];
#pragma unroll
    for (int kt = 0; kt < 4; ++kt) {
        const int koff = kt * 32 + hi * 16;
        const v8h lo = *(const v8h*)&xr[koff];
        const v8h hh = *(const v8h*)&xr[koff + 8];
#pragma unroll
        for (int i = 0; i < 8; ++i) {
            afrag[kt][i]     = lo[i];
            afrag[kt][8 + i] = hh[i];
        }
    }

    // f32 A fragments for the gate GEMM (16x16x4): lane holds row mrow,
    // K = kt*4 + 2*hi + {0,1}  -> one aligned float2 LDS load per chunk.
    v2f gA[4];
#pragma unroll
    for (int kt = 0; kt < 4; ++kt)
        gA[kt] = *(const v2f*)&sF[mrow * 16 + kt * 4 + 2 * hi];

#pragma unroll
    for (int nt = 0; nt < 8; ++nt) {
        const int n = nt * 16 + mrow;

        // message GEMM tile
        v8f c = {};
#pragma unroll
        for (int kt = 0; kt < 4; ++kt) {
            const v16h b = *(const v16h*)&sW[n * F_DIM + kt * 32 + hi * 16];
            c = __builtin_amdgcn_wmma_f32_16x16x32_f16(
                    false, afrag[kt], false, b, (short)0, c, false, false);
        }

        // gate GEMM tile: g = f_ij @ Wg  (K=16 via 4x 16x16x4 f32 WMMA)
        v8f g = {};
#pragma unroll
        for (int kt = 0; kt < 4; ++kt) {
            v2f gB;
            gB[0] = sWg[(kt * 4 + 2 * hi + 0) * F_DIM + n];
            gB[1] = sWg[(kt * 4 + 2 * hi + 1) * F_DIM + n];
            g = __builtin_amdgcn_wmma_f32_16x16x4_f32(
                    false, gA[kt], false, gB, (short)0, g, false, false);
        }

        const float bn = bj[n];
#pragma unroll
        for (int r = 0; r < 8; ++r) {
            const int el = wave * 16 + r + hi * 8;   // local edge for D row
            const int ia = sIi[el];
            if (ia >= 0) {
                const float m = softplus_f(c[r] + bn) * g[r];
                __hip_atomic_fetch_add(&u[(size_t)ia * F_DIM + n], m,
                                       __ATOMIC_RELAXED, __HIP_MEMORY_SCOPE_AGENT);
            }
        }
    }
}

// ------------------------- output head: pred = x @ Wo + bo ------------------
__global__ __launch_bounds__(256)
void head_kernel(const float* __restrict__ X, const float* __restrict__ Wo,
                 const float* __restrict__ bo, float* __restrict__ pred, int nAtoms) {
    int t = blockIdx.x * blockDim.x + threadIdx.x;
    if (t >= nAtoms * 2) return;
    int atom = t >> 1, p = t & 1;
    const float* xr = X + (size_t)atom * F_DIM;
    float acc = bo[p];
#pragma unroll 8
    for (int k = 0; k < F_DIM; ++k) acc = fmaf(xr[k], Wo[k * 2 + p], acc);
    pred[atom * 2 + p] = acc;
}

// ---------------------------------------------------------------------------
extern "C" void kernel_launch(void* const* d_in, const int* in_sizes, int n_in,
                              void* d_out, int out_size, void* d_ws, size_t ws_size,
                              hipStream_t stream) {
    const float* emb  = (const float*)d_in[0];
    const float* f_ij = (const float*)d_in[1];
    const int*   pairs= (const int*)d_in[2];   // pair_indices [2,E] (int32)
    const float* Wg   = (const float*)d_in[3];
    const float* Wi   = (const float*)d_in[4];
    const float* bi   = (const float*)d_in[5];
    const float* Wj   = (const float*)d_in[6];
    const float* bj   = (const float*)d_in[7];
    const float* Wv   = (const float*)d_in[8];
    const float* bv   = (const float*)d_in[9];
    const float* gate = (const float*)d_in[10];
    const float* iW1  = (const float*)d_in[11];
    const float* ib1  = (const float*)d_in[12];
    const float* iW2  = (const float*)d_in[13];
    const float* ib2  = (const float*)d_in[14];
    const float* oW1  = (const float*)d_in[15];
    const float* ob1  = (const float*)d_in[16];
    const float* oW2  = (const float*)d_in[17];
    const float* ob2  = (const float*)d_in[18];
    const float* Wo   = (const float*)d_in[19];
    const float* bo   = (const float*)d_in[20];

    const int N = in_sizes[0] / F_DIM;
    const int E = in_sizes[1] / 16;

    // workspace carve-up
    float* sp = (float*)d_ws;                          // [N,128] softplus(emb) f32
    float* u  = sp + (size_t)N * F_DIM;                // [N,128]
    float* xb = u  + (size_t)N * F_DIM;                // [N,128] output-res state
    _Float16* sph = (_Float16*)(xb + (size_t)N * F_DIM); // [N,128] softplus(emb) f16
    _Float16* tbh = sph + (size_t)N * F_DIM;           // [N,128] residual temp f16
    _Float16* wts = tbh + (size_t)N * F_DIM;
    const int MS = F_DIM * F_DIM;                      // elems per matrix
    _Float16* WiT  = wts;
    _Float16* WjT  = wts + 1  * MS;
    _Float16* WvT  = wts + 2  * MS;
    _Float16* iW1T = wts + 3  * MS;   // 3 matrices
    _Float16* iW2T = wts + 6  * MS;   // 3 matrices
    _Float16* oW1T = wts + 9  * MS;   // 2 matrices
    _Float16* oW2T = wts + 11 * MS;   // 2 matrices

    float* pred = (float*)d_out;              // [N,2]
    float* vout = pred + (size_t)N * 2;       // [N,128] second tuple output

    const int nb = (N + 127) / 128;           // node GEMM blocks
    const int eb = (E + 127) / 128;           // edge blocks

    // 1) sp/sph = softplus(emb)
    {
        size_t n = (size_t)N * F_DIM;
        int blocks = (int)((n + 255) / 256);
        softplus_dual_kernel<<<blocks, 256, 0, stream>>>(emb, sp, sph, n);
    }
    // 2) transpose+convert all 128x128 weights to f16
    wt_transpose_kernel<<<1, 256, 0, stream>>>(Wi,  WiT);
    wt_transpose_kernel<<<1, 256, 0, stream>>>(Wj,  WjT);
    wt_transpose_kernel<<<1, 256, 0, stream>>>(Wv,  WvT);
    wt_transpose_kernel<<<3, 256, 0, stream>>>(iW1, iW1T);
    wt_transpose_kernel<<<3, 256, 0, stream>>>(iW2, iW2T);
    wt_transpose_kernel<<<2, 256, 0, stream>>>(oW1, oW1T);
    wt_transpose_kernel<<<2, 256, 0, stream>>>(oW2, oW2T);

    // 3) u = softplus(sp @ Wi + bi)
    node_gemm_kernel<0, 1, 0><<<nb, 256, 0, stream>>>(sp, WiT, bi, nullptr, nullptr, u, N);

    // 4) u[idx_i] += softplus(sph[idx_j] @ Wj + bj) * (f_ij @ Wg)
    edge_kernel<<<eb, 256, 0, stream>>>(sph, f_ij, pairs, pairs + E, WjT, bj, Wg, u, E);

    // 5) interaction residual blocks: t = sp(sp(u)@W1+b1) ; u += t@W2+b2
    for (int k = 0; k < 3; ++k) {
        node_gemm_kernel<1, 4, 0><<<nb, 256, 0, stream>>>(u, iW1T + k * MS, ib1 + k * F_DIM,
                                                          nullptr, nullptr, tbh, N);
        node_gemm_kernel<0, 2, 1><<<nb, 256, 0, stream>>>(tbh, iW2T + k * MS, ib2 + k * F_DIM,
                                                          u, nullptr, u, N);
    }

    // 6) v = gate * sp + softplus(u) @ Wv + bv   (written straight into d_out)
    node_gemm_kernel<1, 3, 0><<<nb, 256, 0, stream>>>(u, WvT, bv, sp, gate, vout, N);

    // 7) output residual blocks on x (starts at v)
    node_gemm_kernel<1, 4, 0><<<nb, 256, 0, stream>>>(vout, oW1T, ob1, nullptr, nullptr, tbh, N);
    node_gemm_kernel<0, 2, 1><<<nb, 256, 0, stream>>>(tbh, oW2T, ob2, vout, nullptr, xb, N);
    node_gemm_kernel<1, 4, 0><<<nb, 256, 0, stream>>>(xb, oW1T + MS, ob1 + F_DIM, nullptr, nullptr, tbh, N);
    node_gemm_kernel<0, 2, 1><<<nb, 256, 0, stream>>>(tbh, oW2T + MS, ob2 + F_DIM, xb, nullptr, xb, N);

    // 8) prediction = x @ Wo + bo
    head_kernel<<<(N * 2 + 255) / 256, 256, 0, stream>>>(xb, Wo, bo, pred, N);
}
